// QueueMemory_89928025243667
// MI455X (gfx1250) — compile-verified, matrix operand order checked by reading, not run on hardware
//
#include <hip/hip_runtime.h>
#include <cstdint>

// ---------------------------------------------------------------------------
// QueueMemory scan for MI455X (gfx1250, wave32).
//  * scan is sequential over b = 0..63; the carried 8MB memory matrix stays
//    L2-resident (192MB L2), mr-dependent work is hoisted into one fused
//    parallel kernel that reads maximum_routes from HBM exactly once
//    (second S-pass hits L2).
//  * cond_q any-reduction streams the memory matrix through LDS with a
//    double-buffered async pipeline (global_load_async_to_lds_b128 +
//    s_wait_asynccnt 1/0) and reduces the per-lane indicator on the matrix
//    unit via an all-ones V_WMMA_F32_16X16X32_F16 (one atomic per wave).
// ---------------------------------------------------------------------------

typedef __attribute__((ext_vector_type(16))) _Float16 v16h;
typedef __attribute__((ext_vector_type(8)))  float    v8f;

constexpr int   B_ = 64, T_ = 128, F_ = 1024, S_ = 512, M_ = 2048;
constexpr float EPS_   = 0.45f;
constexpr float SQRT_S = 22.62741699796952f; // sqrt(512)

struct Ctrl { float c; float imp; int li; int mi; };

__device__ __forceinline__ float hsig(float x) {
  return fminf(fmaxf(0.2f * x + 0.5f, 0.0f), 1.0f);
}

// Wave-level indicator sum via the matrix unit: D = A x Ones, so the sum of
// all 512 A elements equals (sum of 8 D regs in lane j) + (same in lane j+16).
// Requires uniform control flow (EXEC all ones).
__device__ __forceinline__ float wave_count(bool any) {
  v16h a = {};
  a[0] = any ? (_Float16)1.0f : (_Float16)0.0f;
  v16h ones;
#pragma unroll
  for (int i = 0; i < 16; ++i) ones[i] = (_Float16)1.0f;
  v8f acc = {};
  acc = __builtin_amdgcn_wmma_f32_16x16x32_f16(
      false, a, false, ones, (short)0, acc, false, false);
  float s8 = acc[0] + acc[1] + acc[2] + acc[3] + acc[4] + acc[5] + acc[6] + acc[7];
  return s8 + __shfl_xor(s8, 16, 32);
}

// Async 16B global->LDS copy (per lane). sbase must be wave-uniform.
__device__ __forceinline__ void async_load16(unsigned lds_addr, unsigned voff,
                                             const float* sbase) {
  asm volatile("global_load_async_to_lds_b128 %0, %1, %2"
               :: "v"(lds_addr), "v"(voff), "s"(sbase) : "memory");
}
__device__ __forceinline__ void wait_async_le1() {
  asm volatile("s_wait_asynccnt 1" ::: "memory");
}
__device__ __forceinline__ void wait_async_le0() {
  asm volatile("s_wait_asynccnt 0" ::: "memory");
}

// ---------------------------------------------------------------- init flags
__global__ void k_init(int* condsflag, int* flagq) {
  int t = threadIdx.x;
  if (t < B_) { condsflag[t] = 0; flagq[t] = 0; }
}

// --------------------- fused phase 1: comp_s -> upd_mem_s/upd_idx_s, cond_s
// Pass 1 over S computes comp_s[b][f] (per-column scalar, kept in register);
// pass 2 re-reads the same column slice (L2 hit) for the max blend.
__global__ __launch_bounds__(256) void k_phase1(
    const float* __restrict__ states, const float* __restrict__ mr,
    float* __restrict__ upd_ms, float* __restrict__ upd_is,
    int* __restrict__ condsflag) {
  const int b = blockIdx.y;
  const int f = blockIdx.x * 256 + threadIdx.x;
  const float sv = states[((size_t)b * T_ + (T_ - 1)) * F_ + f];
  const float* mrb = mr + (size_t)b * S_ * F_;

  float ma = 0.0f;
  bool any = false;
  for (int s = 0; s < S_; ++s) {
    float comp = 0.5f - hsig(mrb[(size_t)s * F_ + f] - sv);
    ma = fmaxf(ma, fabsf(comp));
    any = any || (comp >= EPS_);
  }
  const float cs = SQRT_S * ma;

  float tot = wave_count(any);                 // uniform flow up to here
  if ((threadIdx.x & 31) == 0 && tot > 0.5f) atomicOr(&condsflag[b], 1);

  float mx = -INFINITY;
  for (int s = 0; s < S_; ++s)
    mx = fmaxf(mx, mrb[(size_t)s * F_ + f] * cs + sv * (1.0f - cs));
  upd_ms[(size_t)b * F_ + f] = mx;

  if (f == F_ - 1) {
    const float r = sv;                                 // states[b,T-1,F-1]
    upd_is[b] = mrb[F_ - 1] * cs + r * (1.0f - cs);     // mr[b,0,F-1]
  }
}

// -------------------------------- per-step: cond_q any over memory (M x F)
// Each block handles 4 memory rows (4 x 1024 floats) with a double-buffered
// async-to-LDS pipeline; a row is exactly one tile so each thread's columns
// are fixed (f = tid*4 + k).
__global__ __launch_bounds__(256) void k_any(
    const float* __restrict__ memwork, const float* __restrict__ states,
    int b, int* __restrict__ flagq) {
  __shared__ __align__(16) float tile[2][1024];
  const int tid = threadIdx.x;
  const float* gsrc = memwork + (size_t)blockIdx.x * 4096;  // uniform -> SGPR
  // Generic LDS pointer: low 32 bits are the LDS byte address on CDNA5
  // (the aperture is identified purely by the high 32 bits).
  const unsigned lds0 = (unsigned)(size_t)(void*)&tile[0][0] + (unsigned)tid * 16u;
  const unsigned lds1 = (unsigned)(size_t)(void*)&tile[1][0] + (unsigned)tid * 16u;
  const unsigned voff = (unsigned)tid * 16u;

  const float* sv = states + ((size_t)b * T_ + (T_ - 1)) * F_;
  float s0 = sv[tid * 4 + 0], s1 = sv[tid * 4 + 1];
  float s2 = sv[tid * 4 + 2], s3 = sv[tid * 4 + 3];

  async_load16(lds0, voff, gsrc);                       // row 0 in flight
  bool any = false;
#pragma unroll
  for (int t = 0; t < 4; ++t) {
    if (t < 3)
      async_load16(((t + 1) & 1) ? lds1 : lds0,
                   voff + (unsigned)(t + 1) * 4096u, gsrc);
    if (t < 3) wait_async_le1(); else wait_async_le0(); // oldest tile done
    const float* tp = tile[t & 1] + tid * 4;
    any = any || ((0.5f - hsig(tp[0] - s0)) >= EPS_);
    any = any || ((0.5f - hsig(tp[1] - s1)) >= EPS_);
    any = any || ((0.5f - hsig(tp[2] - s2)) >= EPS_);
    any = any || ((0.5f - hsig(tp[3] - s3)) >= EPS_);
  }

  float tot = wave_count(any);                          // uniform flow
  if ((tid & 31) == 0 && tot > 0.5f) atomicOr(&flagq[b], 1);
}

// ----------------- per-step control kernel: cq stats + full index machine
__global__ __launch_bounds__(256) void k_t1(
    float* __restrict__ memwork, float* __restrict__ idxwork,
    const float* __restrict__ states, const int* __restrict__ flagq,
    const int* __restrict__ condsflag, const float* __restrict__ upd_is,
    int* __restrict__ sel, float* __restrict__ row_mi,
    Ctrl* __restrict__ ctrl, int b) {
  __shared__ float s_cq[M_];
  __shared__ float s_i1[M_];
  __shared__ float s_i2[M_];
  __shared__ float red[256];
  __shared__ int   redi[256];
  const int tid = threadIdx.x;
  const float* sv = states + ((size_t)b * T_ + (T_ - 1)) * F_;
  const float s_last = sv[F_ - 1];
  const float r = s_last;

  for (int m = tid; m < M_; m += 256)
    s_cq[m] = 0.5f - hsig(memwork[(size_t)m * F_ + (F_ - 1)] - s_last);
  __syncthreads();

  // c = max(cq)
  float lm = -INFINITY;
  for (int m = tid; m < M_; m += 256) lm = fmaxf(lm, s_cq[m]);
  red[tid] = lm; __syncthreads();
  for (int o = 128; o > 0; o >>= 1) { if (tid < o) red[tid] = fmaxf(red[tid], red[tid + o]); __syncthreads(); }
  const float c = red[0]; __syncthreads();

  // mi = first argmax(cq)
  int lmi = M_;
  for (int m = tid; m < M_; m += 256) if (s_cq[m] == c && m < lmi) lmi = m;
  redi[tid] = lmi; __syncthreads();
  for (int o = 128; o > 0; o >>= 1) { if (tid < o) redi[tid] = min(redi[tid], redi[tid + o]); __syncthreads(); }
  const int mi = redi[0]; __syncthreads();

  const bool condq = (flagq[b] != 0);
  const bool conds = (condsflag[b] != 0);
  const bool cond_e = !condq && !conds;
  const float uis = upd_is[b];
  const float upd_idx_q = idxwork[mi] * (1.0f - c) + r * c;

  // idx1 + max(-idx1)
  float ln = -INFINITY;
  for (int m = tid; m < M_; m += 256) {
    float iold = idxwork[m];
    bool  mq = (s_cq[m] == c);
    float i1 = condq ? (mq ? iold : upd_idx_q) : iold;
    s_i1[m] = i1;
    ln = fmaxf(ln, -i1);
  }
  red[tid] = ln; __syncthreads();
  for (int o = 128; o > 0; o >>= 1) { if (tid < o) red[tid] = fmaxf(red[tid], red[tid + o]); __syncthreads(); }
  const float maxneg1 = red[0]; __syncthreads();

  // idx2 + max(-idx2)
  ln = -INFINITY;
  for (int m = tid; m < M_; m += 256) {
    float i1 = s_i1[m];
    bool  mm1 = (i1 == maxneg1);
    float i2 = conds ? (mm1 ? uis : i1) : i1;
    s_i2[m] = i2;
    ln = fmaxf(ln, -i2);
  }
  red[tid] = ln; __syncthreads();
  for (int o = 128; o > 0; o >>= 1) { if (tid < o) red[tid] = fmaxf(red[tid], red[tid + o]); __syncthreads(); }
  const float maxneg2 = red[0]; __syncthreads();

  // idx3, per-row memory selector (0=keep 1=blend(mi,s) 2=upd_mem_s 3=s)
  for (int m = tid; m < M_; m += 256) {
    float i2 = s_i2[m];
    bool  mm2 = (i2 == maxneg2);
    float i3 = cond_e ? (mm2 ? r : i2) : i2;
    idxwork[m] = i3;
    int sl = 0;
    if (condq && !(s_cq[m] == c)) sl = 1;
    if (conds && (s_i1[m] == maxneg1)) sl = 2;
    if (cond_e && mm2) sl = 3;
    sel[m] = sl;
    s_i2[m] = i3;
  }
  __syncthreads();

  // li = first argmax(idx3)
  lm = -INFINITY;
  for (int m = tid; m < M_; m += 256) lm = fmaxf(lm, s_i2[m]);
  red[tid] = lm; __syncthreads();
  for (int o = 128; o > 0; o >>= 1) { if (tid < o) red[tid] = fmaxf(red[tid], red[tid + o]); __syncthreads(); }
  const float i3max = red[0]; __syncthreads();
  lmi = M_;
  for (int m = tid; m < M_; m += 256) if (s_i2[m] == i3max && m < lmi) lmi = m;
  redi[tid] = lmi; __syncthreads();
  for (int o = 128; o > 0; o >>= 1) { if (tid < o) redi[tid] = min(redi[tid], redi[tid + o]); __syncthreads(); }
  const int li = redi[0];

  // snapshot old memory row mi (k_apply may overwrite it)
  for (int f = tid; f < F_; f += 256) row_mi[f] = memwork[(size_t)mi * F_ + f];
  if (tid == 0) { ctrl->c = c; ctrl->mi = mi; ctrl->li = li; ctrl->imp = s_i2[li]; }
}

// --------------------------------------------- per-step memory select/write
__global__ __launch_bounds__(256) void k_apply(
    float* __restrict__ memwork, const int* __restrict__ sel,
    const float* __restrict__ row_mi, const float* __restrict__ upd_ms_b,
    const float* __restrict__ sv, const Ctrl* __restrict__ ctrl) {
  const int m = blockIdx.y;
  const int f = blockIdx.x * 256 + threadIdx.x;
  const int sl = sel[m];
  if (sl == 0) return;
  float v;
  if (sl == 1)      { float c = ctrl->c; v = row_mi[f] * (1.0f - c) + sv[f] * c; }
  else if (sl == 2) { v = upd_ms_b[f]; }
  else              { v = sv[f]; }
  memwork[(size_t)m * F_ + f] = v;
}

// --------------------------------------------------- per-step leader output
__global__ __launch_bounds__(256) void k_t2(
    const float* __restrict__ memwork, const Ctrl* __restrict__ ctrl,
    float* __restrict__ out_t, float* __restrict__ out_i, int b) {
  const int f = blockIdx.x * 256 + threadIdx.x;
  const int li = ctrl->li;
  out_t[(size_t)b * F_ + f] = memwork[(size_t)li * F_ + f];
  if (f == 0) out_i[b] = ctrl->imp;
}

// ---------------------------------------------------------------------------
extern "C" void kernel_launch(void* const* d_in, const int* in_sizes, int n_in,
                              void* d_out, int out_size, void* d_ws, size_t ws_size,
                              hipStream_t stream) {
  (void)in_sizes; (void)n_in; (void)out_size; (void)ws_size;
  const float* states = (const float*)d_in[0];   // (64,128,1024)
  const float* mr     = (const float*)d_in[1];   // (64,512,1024)
  const float* memory = (const float*)d_in[2];   // (2048,1024)
  const float* index  = (const float*)d_in[3];   // (2048,1)

  char* ws = (char*)d_ws;
  size_t off = 0;
  auto carve = [&](size_t bytes) -> void* {
    void* p = ws + off; off += (bytes + 255) & ~(size_t)255; return p;
  };
  float* memwork  = (float*)carve((size_t)M_ * F_ * 4);   // carried memory
  float* idxwork  = (float*)carve((size_t)M_ * 4);        // carried index
  float* upd_ms   = (float*)carve((size_t)B_ * F_ * 4);
  float* upd_is   = (float*)carve((size_t)B_ * 4);
  int*   condsflg = (int*)carve((size_t)B_ * 4);
  int*   flagq    = (int*)carve((size_t)B_ * 4);
  int*   sel      = (int*)carve((size_t)M_ * 4);
  float* row_mi   = (float*)carve((size_t)F_ * 4);
  Ctrl*  ctrl     = (Ctrl*)carve(sizeof(Ctrl));

  float* out_t = (float*)d_out;                 // targets (64,1,1024)
  float* out_i = out_t + (size_t)B_ * F_;       // importances (64,1,1)

  hipMemcpyAsync(memwork, memory, (size_t)M_ * F_ * 4, hipMemcpyDeviceToDevice, stream);
  hipMemcpyAsync(idxwork, index,  (size_t)M_ * 4,      hipMemcpyDeviceToDevice, stream);
  k_init<<<1, 128, 0, stream>>>(condsflg, flagq);

  dim3 g1(F_ / 256, B_);
  k_phase1<<<g1, 256, 0, stream>>>(states, mr, upd_ms, upd_is, condsflg);

  for (int b = 0; b < B_; ++b) {
    const float* sv = states + ((size_t)b * T_ + (T_ - 1)) * F_;
    k_any<<<(M_ * F_) / 4096, 256, 0, stream>>>(memwork, states, b, flagq);
    k_t1<<<1, 256, 0, stream>>>(memwork, idxwork, states, flagq, condsflg,
                                upd_is, sel, row_mi, ctrl, b);
    dim3 ga(F_ / 256, M_);
    k_apply<<<ga, 256, 0, stream>>>(memwork, sel, row_mi,
                                    upd_ms + (size_t)b * F_, sv, ctrl);
    k_t2<<<F_ / 256, 256, 0, stream>>>(memwork, ctrl, out_t, out_i, b);
  }
}
